// RNNAgent_5987184410730
// MI455X (gfx1250) — matrix operand order repaired and checked by reading, not compiled, not run on hardware
//
#include <hip/hip_runtime.h>

// ---------------------------------------------------------------------------
// RNNAgent forward for MI455X (gfx1250, wave32, WMMA + async-to-LDS DMA).
//
// Phase A (parallel, fills GPU): fc_in (128->64->64) + hoisted GRU input gates
//          gi = W_ih * x2 + b_ih   -> workspace, f16, [n][bt][t][16][192]
//          x-tiles double-buffered via global_load_async_to_lds_b128.
// Phase B (serial over T): gh = W_hh*h + gates + fused fc_out -> d_out,
//          gi tile for t+1 DMA'd to LDS while step t computes (ASYNCcnt).
// All GEMMs use v_wmma_f32_16x16x32_f16 (f32 accumulate).
// ---------------------------------------------------------------------------

#define BB   64
#define NN   4
#define TT   1024
#define OBSF 100
#define ADIM 28
#define FF   128
#define HH   64
#define H2   32
#define G3   192            // 3*H
#define TCHUNK 16

typedef __attribute__((ext_vector_type(16))) _Float16 v16h;
typedef __attribute__((ext_vector_type(8)))  _Float16 v8h;
typedef __attribute__((ext_vector_type(8)))  float    v8f;

__device__ __forceinline__ v8f vzero() {
    v8f z;
#pragma unroll
    for (int i = 0; i < 8; ++i) z[i] = 0.0f;
    return z;
}

// LDS byte address: generic pointers to LDS carry the LDS offset in addr[31:0].
__device__ __forceinline__ unsigned lds_addr32(const void* p) {
    return (unsigned)(unsigned long long)p;
}

// Async DMA: global -> LDS, 16 bytes per lane, tracked by ASYNCcnt.
__device__ __forceinline__ void async_ld_b128(unsigned lds_off, unsigned long long gaddr) {
    asm volatile("global_load_async_to_lds_b128 %0, %1, off"
                 :: "v"(lds_off), "v"(gaddr) : "memory");
}

// A-fragment, f16 16x32 (MxK). ISA layout: lanes 0-15 rows M=0..15 hold K
// runs [k0..k0+7] and [k0+16..k0+23]; lanes 16-31 hold [k0+8..15],[k0+24..31].
__device__ __forceinline__ v16h frag_a16(const _Float16* base, int stride, int k0, int lane) {
    const _Float16* p = base + (lane & 15) * stride + k0 + ((lane >> 4) << 3);
    v8h lo = *(const v8h*)p;
    v8h hi = *(const v8h*)(p + 16);
    return __builtin_shufflevector(lo, hi, 0,1,2,3,4,5,6,7,8,9,10,11,12,13,14,15);
}

// B-fragment, f16 32x16 (KxN), LDS tile stored [col][K] (== weight [out][in]).
__device__ __forceinline__ v16h frag_b16(const _Float16* base, int stride, int k0, int lane) {
    const _Float16* p = base + (lane & 15) * stride + k0 + ((lane >> 4) << 3);
    v8h lo = *(const v8h*)p;
    v8h hi = *(const v8h*)(p + 16);
    return __builtin_shufflevector(lo, hi, 0,1,2,3,4,5,6,7,8,9,10,11,12,13,14,15);
}

__device__ __forceinline__ v8f wmma16(v16h a, v16h b, v8f c) {
    return __builtin_amdgcn_wmma_f32_16x16x32_f16(false, a, false, b, (short)0, c, false, false);
}

__device__ __forceinline__ float sigmoidf_(float x) { return 1.0f / (1.0f + __expf(-x)); }

// ---------------------------------------------------------------------------
// Kernel A: fc_in + gi.  Grid: NN * 4 * (TT/TCHUNK) blocks, 128 threads.
// ---------------------------------------------------------------------------

// Issue the 512 16B chunks (4/thread) of the raw f32 x-tile for timestep t.
__device__ __forceinline__ void issue_x_tile(
    int tid, int b0, int n, int t,
    const float* __restrict__ obs, const float* __restrict__ act,
    unsigned lds_raw /* byte addr of raw buffer */)
{
#pragma unroll
    for (int q4 = 0; q4 < 4; ++q4) {
        const int q = tid + q4 * 128;
        unsigned long long g;
        unsigned l;
        if (q < 400) {                       // obs rows: 16 x 400B
            const int r = q / 25, c = q % 25;
            g = (unsigned long long)(const void*)
                    (obs + ((long)((b0 + r) * NN + n) * TT + t) * OBSF) + c * 16;
            l = lds_raw + r * 512 + c * 16;
        } else {                             // action rows: 16 x 112B
            const int qa = q - 400;
            const int r = qa / 7, c = qa % 7;
            g = (unsigned long long)(const void*)
                    (act + ((long)((b0 + r) * NN + n) * TT + t) * ADIM) + c * 16;
            l = lds_raw + r * 512 + 400 + c * 16;
        }
        async_ld_b128(l, g);
    }
}

__global__ __launch_bounds__(128) void fcin_gi_kernel(
    const float* __restrict__ obs, const float* __restrict__ act,
    const float* __restrict__ W1,  const float* __restrict__ b1,
    const float* __restrict__ W2,  const float* __restrict__ b2,
    const float* __restrict__ Wih, const float* __restrict__ bih,
    _Float16* __restrict__ gi)
{
    __shared__ __align__(32) _Float16 sW1 [HH * FF];      // 16 KB  [h][f]
    __shared__ __align__(32) _Float16 sW2 [HH * HH];      //  8 KB  [g][h]
    __shared__ __align__(32) _Float16 sWih[G3 * HH];      // 24 KB  [k][h]
    __shared__ __align__(32) float    sXraw[2 * 16 * FF]; // 16 KB double buffer
    __shared__ __align__(32) _Float16 sX  [16 * FF];      //  4 KB
    __shared__ __align__(32) _Float16 sH1 [16 * HH];      //  2 KB
    __shared__ __align__(32) _Float16 sH2 [16 * HH];      //  2 KB
    __shared__ __align__(32) _Float16 sGO [16 * G3];      //  6 KB gi staging

    const int tid  = threadIdx.x;
    const int lane = tid & 31;
    const int wave = tid >> 5;
    const int ntc  = TT / TCHUNK;
    const int wg   = blockIdx.x;
    const int n    = wg / (4 * ntc);
    const int bt   = (wg / ntc) & 3;
    const int tc   = wg % ntc;
    const int b0   = bt * 16;

    // Stage weights once (f32 -> f16)
    for (int e = tid; e < HH * FF; e += 128) sW1[e]  = (_Float16)W1 [n * HH * FF + e];
    for (int e = tid; e < HH * HH; e += 128) sW2[e]  = (_Float16)W2 [n * HH * HH + e];
    for (int e = tid; e < G3 * HH; e += 128) sWih[e] = (_Float16)Wih[n * G3 * HH + e];

    const int col    = lane & 15;
    const int rowoff = (lane >> 4) * 8;
    const float bias1 = b1[n * HH + wave * 16 + col];
    const float bias2 = b2[n * HH + wave * 16 + col];
    float biih[3];
#pragma unroll
    for (int j = 0; j < 3; ++j) biih[j] = bih[n * G3 + (wave * 3 + j) * 16 + col];

    // Prologue: DMA the first x-tile into raw buffer 0.
    issue_x_tile(tid, b0, n, tc * TCHUNK, obs, act, lds_addr32(sXraw));

    for (int tt = 0; tt < TCHUNK; ++tt) {
        const int t   = tc * TCHUNK + tt;
        const int cur = tt & 1;

        // Kick the DMA for t+1 into the other raw buffer, then wait for the
        // 4 loads of the *previous* batch (in-order completion).
        const int tnext = (t + 1 < TT) ? t + 1 : t;
        issue_x_tile(tid, b0, n, tnext, obs, act,
                     lds_addr32(sXraw + (cur ^ 1) * 16 * FF));
        asm volatile("s_wait_asynccnt 4" ::: "memory");
        __syncthreads();

        // Convert raw f32 tile -> f16 WMMA A-tile.
        const float* raw = sXraw + cur * 16 * FF;
        for (int e = tid; e < 16 * FF; e += 128) sX[e] = (_Float16)raw[e];
        __syncthreads();

        // fc_in layer 1: [16x128] x [128x64] -> relu -> sH1. Wave w: cols w*16..
        v8f acc = vzero();
#pragma unroll
        for (int kk = 0; kk < 4; ++kk)
            acc = wmma16(frag_a16(sX, FF, kk * 32, lane),
                         frag_b16(sW1 + wave * 16 * FF, FF, kk * 32, lane), acc);
#pragma unroll
        for (int r = 0; r < 8; ++r) {
            float v = acc[r] + bias1; v = v > 0.f ? v : 0.f;
            sH1[(r + rowoff) * HH + wave * 16 + col] = (_Float16)v;
        }
        __syncthreads();

        // fc_in layer 2: [16x64] x [64x64] -> relu -> sH2
        acc = vzero();
#pragma unroll
        for (int kk = 0; kk < 2; ++kk)
            acc = wmma16(frag_a16(sH1, HH, kk * 32, lane),
                         frag_b16(sW2 + wave * 16 * HH, HH, kk * 32, lane), acc);
#pragma unroll
        for (int r = 0; r < 8; ++r) {
            float v = acc[r] + bias2; v = v > 0.f ? v : 0.f;
            sH2[(r + rowoff) * HH + wave * 16 + col] = (_Float16)v;
        }
        __syncthreads();

        // gi = x2 @ W_ih^T + b_ih : [16x64] x [64x192]. Wave w: 3 col-tiles.
        v16h a0 = frag_a16(sH2, HH, 0, lane);
        v16h a1 = frag_a16(sH2, HH, 32, lane);
#pragma unroll
        for (int j = 0; j < 3; ++j) {
            const int nt = wave * 3 + j;
            v8f accg = vzero();
            accg = wmma16(a0, frag_b16(sWih + nt * 16 * HH, HH, 0,  lane), accg);
            accg = wmma16(a1, frag_b16(sWih + nt * 16 * HH, HH, 32, lane), accg);
#pragma unroll
            for (int r = 0; r < 8; ++r)
                sGO[(r + rowoff) * G3 + nt * 16 + col] = (_Float16)(accg[r] + biih[j]);
        }
        __syncthreads();

        // Coalesced 16B-chunk store of the gi tile (384 chunks, 3/thread).
        {
            _Float16* gout = gi + ((n * 4 + bt) * TT + t) * (16 * G3);
            const v8h* src = (const v8h*)sGO;
            v8h*       dst = (v8h*)gout;
#pragma unroll
            for (int q = 0; q < 3; ++q) dst[tid + q * 128] = src[tid + q * 128];
        }
        __syncthreads();
    }
}

// ---------------------------------------------------------------------------
// Kernel B: GRU recurrence + fused fc_out. Grid: 16 blocks (n, b-tile), 128 thr.
// ---------------------------------------------------------------------------
__global__ __launch_bounds__(128) void gru_fcout_kernel(
    const float* __restrict__ h0,
    const float* __restrict__ Whh, const float* __restrict__ bhh,
    const float* __restrict__ Wo1, const float* __restrict__ bo1,
    const float* __restrict__ Wo2, const float* __restrict__ bo2,
    const _Float16* __restrict__ gi, float* __restrict__ out)
{
    __shared__ __align__(32) _Float16 sWhh[G3 * HH];     // 24 KB [k][h]
    __shared__ __align__(32) _Float16 sWo1[H2 * HH];     //  4 KB
    __shared__ __align__(32) _Float16 sWo2[32 * H2];     //  2 KB (rows 28..31 = 0)
    __shared__ __align__(32) _Float16 sGI [2 * 16 * G3]; // 12 KB gi double buffer
    __shared__ __align__(32) _Float16 sHh [16 * HH];     // h, f16 (WMMA A)
    __shared__ __align__(32) float    sHf [16 * HH];     // h, f32 (gate math)
    __shared__ __align__(32) float    sARZ[16 * 128];    // gi+gh+b for r,z cols
    __shared__ __align__(32) float    sGIN[16 * HH];     // i_n
    __shared__ __align__(32) float    sGHN[16 * HH];     // h_n + b_hn
    __shared__ __align__(32) _Float16 sO1 [16 * H2];

    const int tid  = threadIdx.x;
    const int lane = tid & 31;
    const int wave = tid >> 5;
    const int n    = blockIdx.x >> 2;
    const int bt   = blockIdx.x & 3;

    for (int e = tid; e < G3 * HH; e += 128) sWhh[e] = (_Float16)Whh[n * G3 * HH + e];
    for (int e = tid; e < H2 * HH; e += 128) sWo1[e] = (_Float16)Wo1[n * H2 * HH + e];
    for (int e = tid; e < 32 * H2; e += 128) {
        const int a = e >> 5, k = e & 31;
        sWo2[e] = (a < ADIM) ? (_Float16)Wo2[(n * ADIM + a) * H2 + k] : (_Float16)0.f;
    }
    for (int e = tid; e < 16 * HH; e += 128) {
        const int r = e >> 6, h = e & 63;
        const float v = h0[((bt * 16 + r) * NN + n) * HH + h];
        sHf[e] = v; sHh[e] = (_Float16)v;
    }

    const int col    = lane & 15;
    const int rowoff = (lane >> 4) * 8;
    float cbhh[3];
#pragma unroll
    for (int j = 0; j < 3; ++j) cbhh[j] = bhh[n * G3 + (wave * 3 + j) * 16 + col];
    const float cbo1 = (wave < 2) ? bo1[n * H2 + wave * 16 + col] : 0.f;
    const float cbo2 = (wave < 2 && (wave * 16 + col) < ADIM)
                           ? bo2[n * ADIM + wave * 16 + col] : 0.f;

    const _Float16* giBase = gi + (long)(n * 4 + bt) * TT * 16 * G3;

    // Prologue: DMA gi tile for t=0 into buffer 0 (384 chunks, 3/thread).
    {
        const unsigned long long g0 = (unsigned long long)(const void*)giBase;
        const unsigned l0 = lds_addr32(sGI);
#pragma unroll
        for (int q = 0; q < 3; ++q) {
            const int byte = (tid + q * 128) * 16;
            async_ld_b128(l0 + byte, g0 + byte);
        }
    }

    for (int t = 0; t < TT; ++t) {
        const int cur = t & 1;

        // Kick DMA for t+1 into the other buffer; wait for the previous batch.
        {
            const int tn = (t + 1 < TT) ? t + 1 : t;
            const unsigned long long gn =
                (unsigned long long)(const void*)(giBase + (long)tn * 16 * G3);
            const unsigned ln = lds_addr32(sGI + (cur ^ 1) * 16 * G3);
#pragma unroll
            for (int q = 0; q < 3; ++q) {
                const int byte = (tid + q * 128) * 16;
                async_ld_b128(ln + byte, gn + byte);
            }
        }
        asm volatile("s_wait_asynccnt 3" ::: "memory");
        __syncthreads();

        const _Float16* gcur = sGI + cur * 16 * G3;

        // gh = h @ W_hh^T (+ b_hh); merge with gi into gate staging.
        v16h a0 = frag_a16(sHh, HH, 0,  lane);
        v16h a1 = frag_a16(sHh, HH, 32, lane);
#pragma unroll
        for (int j = 0; j < 3; ++j) {
            const int nt = wave * 3 + j;
            v8f acc = vzero();
            acc = wmma16(a0, frag_b16(sWhh + nt * 16 * HH, HH, 0,  lane), acc);
            acc = wmma16(a1, frag_b16(sWhh + nt * 16 * HH, HH, 32, lane), acc);
            const int c = nt * 16 + col;
#pragma unroll
            for (int r = 0; r < 8; ++r) {
                const int row = r + rowoff;
                const float ghv = acc[r] + cbhh[j];
                const float giv = (float)gcur[row * G3 + c];
                if (c < 128) {
                    sARZ[row * 128 + c] = giv + ghv;           // r,z pre-acts
                } else {
                    sGIN[row * HH + (c - 128)] = giv;           // i_n
                    sGHN[row * HH + (c - 128)] = ghv;           // h_n + b_hn
                }
            }
        }
        __syncthreads();

        // Gates: h' = (1-z)*tanh(i_n + r*(h_n+b)) + z*h   (8 elems / thread)
#pragma unroll
        for (int i = 0; i < 8; ++i) {
            const int e = tid + i * 128;
            const int row = e >> 6, h = e & 63;
            const float rg = sigmoidf_(sARZ[row * 128 + h]);
            const float zg = sigmoidf_(sARZ[row * 128 + 64 + h]);
            const float nv = tanhf(sGIN[row * HH + h] + rg * sGHN[row * HH + h]);
            const float hn = (1.f - zg) * nv + zg * sHf[e];
            sHf[e] = hn; sHh[e] = (_Float16)hn;
        }
        __syncthreads();

        // fc_out layer 1: o1 = relu(h' @ Wo1^T + bo1)  (waves 0,1)
        if (wave < 2) {
            v8f acc = vzero();
            acc = wmma16(frag_a16(sHh, HH, 0,  lane),
                         frag_b16(sWo1 + wave * 16 * HH, HH, 0,  lane), acc);
            acc = wmma16(frag_a16(sHh, HH, 32, lane),
                         frag_b16(sWo1 + wave * 16 * HH, HH, 32, lane), acc);
#pragma unroll
            for (int r = 0; r < 8; ++r) {
                float v = acc[r] + cbo1; v = v > 0.f ? v : 0.f;
                sO1[(r + rowoff) * H2 + wave * 16 + col] = (_Float16)v;
            }
        }
        __syncthreads();

        // fc_out layer 2: q = o1 @ Wo2^T + bo2 -> out[B][T][N][A]
        if (wave < 2) {
            v8f acc = vzero();
            acc = wmma16(frag_a16(sO1, H2, 0, lane),
                         frag_b16(sWo2 + wave * 16 * H2, H2, 0, lane), acc);
            const int c = wave * 16 + col;
            if (c < ADIM) {
#pragma unroll
                for (int r = 0; r < 8; ++r) {
                    const int b = bt * 16 + r + rowoff;
                    out[(((long)b * TT + t) * NN + n) * ADIM + c] = acc[r] + cbo2;
                }
            }
        }
        __syncthreads();
    }
}

// ---------------------------------------------------------------------------
extern "C" void kernel_launch(void* const* d_in, const int* in_sizes, int n_in,
                              void* d_out, int out_size, void* d_ws, size_t ws_size,
                              hipStream_t stream) {
    const float* obs  = (const float*)d_in[0];
    const float* act  = (const float*)d_in[1];
    const float* h0   = (const float*)d_in[2];
    const float* W1   = (const float*)d_in[3];
    const float* b1   = (const float*)d_in[4];
    const float* W2   = (const float*)d_in[5];
    const float* b2   = (const float*)d_in[6];
    const float* Wih  = (const float*)d_in[7];
    const float* Whh  = (const float*)d_in[8];
    const float* bih  = (const float*)d_in[9];
    const float* bhh  = (const float*)d_in[10];
    const float* Wo1  = (const float*)d_in[11];
    const float* bo1  = (const float*)d_in[12];
    const float* Wo2  = (const float*)d_in[13];
    const float* bo2  = (const float*)d_in[14];
    float* out = (float*)d_out;

    _Float16* gi = (_Float16*)d_ws;   // [N][4][T][16][192] f16 = 96 MB

    fcin_gi_kernel<<<NN * 4 * (TT / TCHUNK), 128, 0, stream>>>(
        obs, act, W1, b1, W2, b2, Wih, bih, gi);
    gru_fcout_kernel<<<NN * 4, 128, 0, stream>>>(
        h0, Whh, bhh, Wo1, bo1, Wo2, bo2, gi, out);
}
